// InteractionFusion_369367187531
// MI455X (gfx1250) — matrix operand order repaired and checked by reading, not compiled, not run on hardware
//
#include <hip/hip_runtime.h>
#include <hip/hip_bf16.h>

typedef __attribute__((ext_vector_type(16))) __bf16          bf16x16;
typedef __attribute__((ext_vector_type(16))) unsigned short  ushort16;
typedef __attribute__((ext_vector_type(8)))  float           floatx8;

#define DEVI static __device__ __forceinline__

constexpr int   N_ = 16, C_ = 256, H_ = 56, W_ = 56, HW_ = H_ * W_;
constexpr int   OUT_ = 256, G_ = 8, KS_ = 56, GP_ = OUT_ / G_;   // GP=32
constexpr int   B_ = N_ * W_;                                    // 896 batches
constexpr int   QKV_ROWS  = 512;                                 // q(128)+k(128)+v(256)
constexpr int   QKV_PITCH = QKV_ROWS * H_;                       // floats per batch
constexpr float EPSF = 1e-5f;

// ---- workspace layout (float offsets) ----
constexpr size_t OFF_PR    = 0;          // 4096
constexpr size_t OFF_PD    = 4096;       // 4096
constexpr size_t OFF_SR    = 8192;       // 4096  (co + ca_r)
constexpr size_t OFF_SD    = 12288;      // 4096  (co + ca_d)
constexpr size_t OFF_QSTAT = 16384;      // 512*2 (mean,rstd)
constexpr size_t OFF_SSTAT = 17408;      // 24*2  (sum,sumsq)  atomically built
constexpr size_t OFF_OSTAT = 17472;      // 512*2 (sum,sumsq)  atomically built
constexpr size_t OFF_WB    = 20480;      // 512*256 bf16 weights (q;k;v)
constexpr size_t OFF_QKV   = 131072;     // 896*512*56 f32
constexpr size_t OFF_SO    = OFF_QKV + (size_t)B_ * QKV_PITCH;   // 896*512*56 f32

DEVI unsigned short f2bf(float f) {              // round-to-nearest-even f32 -> bf16
    unsigned u = __builtin_bit_cast(unsigned, f);
    u += 0x7fffu + ((u >> 16) & 1u);
    return (unsigned short)(u >> 16);
}

DEVI float blk_sum(float v, float* red) {
    int t = threadIdx.x;
    red[t] = v; __syncthreads();
    for (int s = blockDim.x >> 1; s > 0; s >>= 1) {
        if (t < s) red[t] += red[t + s];
        __syncthreads();
    }
    float r = red[0]; __syncthreads();
    return r;
}
DEVI float blk_max(float v, float* red) {
    int t = threadIdx.x;
    red[t] = v; __syncthreads();
    for (int s = blockDim.x >> 1; s > 0; s >>= 1) {
        if (t < s) red[t] = fmaxf(red[t], red[t + s]);
        __syncthreads();
    }
    float r = red[0]; __syncthreads();
    return r;
}

// ------------------------------------------------------------------ zero stats
__global__ void k_zero(float* ws) {
    int t = threadIdx.x;
    for (int i = t; i < 48;   i += 256) ws[OFF_SSTAT + i] = 0.0f;
    for (int i = t; i < 1024; i += 256) ws[OFF_OSTAT + i] = 0.0f;
}

// ------------------------------------------------------------- global avg pool
__global__ void k_pool(const float* __restrict__ x, const float* __restrict__ xd,
                       float* ws) {
    __shared__ float red[256];
    int nc = blockIdx.x;                        // n*256 + c
    const float* px = x  + (size_t)nc * HW_;
    const float* pd = xd + (size_t)nc * HW_;
    float s = 0.f, sd = 0.f;
    for (int i = threadIdx.x; i < HW_; i += 256) { s += px[i]; sd += pd[i]; }
    s  = blk_sum(s,  red);
    sd = blk_sum(sd, red);
    if (threadIdx.x == 0) {
        ws[OFF_PR + nc] = s  * (1.0f / HW_);
        ws[OFF_PD + nc] = sd * (1.0f / HW_);
    }
}

// ------------------------------------------------- channel attention + softmax
__global__ void k_chatt(const float* __restrict__ wr, const float* __restrict__ br,
                        const float* __restrict__ wd, const float* __restrict__ bd,
                        float* ws) {
    __shared__ float red[256];
    int n = blockIdx.x, o = threadIdx.x;
    const float* pr = ws + OFF_PR + n * C_;
    const float* pd = ws + OFF_PD + n * C_;
    float ar = br[o], ad = bd[o];
    for (int c = 0; c < C_; ++c) {
        ar += pr[c] * wr[o * C_ + c];
        ad += pd[c] * wd[o * C_ + c];
    }
    ar = (ar > 0.f) ? 1.f / (1.f + __expf(-ar)) : 0.5f;   // sigmoid(relu(x))
    ad = (ad > 0.f) ? 1.f / (1.f + __expf(-ad)) : 0.5f;
    float t  = ar + ad;
    float mx = blk_max(t, red);
    float e  = __expf(t - mx);
    float se = blk_sum(e, red);
    float co = e / se;
    ws[OFF_SR + n * C_ + o] = co + ar;
    ws[OFF_SD + n * C_ + o] = co + ad;
}

// ---------------------------------------------------------- weights -> bf16
__global__ void k_wcvt(const float* __restrict__ qw, const float* __restrict__ kw,
                       const float* __restrict__ vw, float* ws) {
    unsigned short* wb = (unsigned short*)(ws + OFF_WB);
    int idx = blockIdx.x * 256 + threadIdx.x;     // 512 blocks -> 131072 elems
    int row = idx >> 8, c = idx & 255;
    float v = (row < 128) ? qw[row * C_ + c]
            : (row < 256) ? kw[(row - 128) * C_ + c]
                          : vw[(row - 256) * C_ + c];
    wb[idx] = f2bf(v);
}

// --------------------------------------------- q/k/v projection GEMMs (WMMA)
// one block per b = n*W + w; 8 waves; activations in dynamic LDS (3 x 64x256 bf16)
__global__ void k_proj(const float* __restrict__ x, const float* __restrict__ xd,
                       float* ws) {
    extern __shared__ __align__(16) unsigned char smem_raw[];
    unsigned short* xbr = (unsigned short*)smem_raw;        // [h64][c256]
    unsigned short* xbd = xbr + 64 * C_;
    unsigned short* xbf = xbd + 64 * C_;

    int b = blockIdx.x, n = b / W_, w = b % W_;
    int t = threadIdx.x;
    const float* sr = ws + OFF_SR + n * C_;
    const float* sd = ws + OFF_SD + n * C_;

    for (int idx = t; idx < 8 * C_; idx += 256) {           // zero pad rows 56..63
        int off = (56 * C_) + idx;
        xbr[off] = 0; xbd[off] = 0; xbf[off] = 0;
    }
    __syncthreads();
    for (int idx = t; idx < H_ * C_; idx += 256) {          // h fastest -> coalesce-ish
        int c = idx / H_, h = idx % H_;
        float xv = x [((size_t)(n * C_ + c)) * HW_ + h * W_ + w];
        float dv = xd[((size_t)(n * C_ + c)) * HW_ + h * W_ + w];
        xbr[h * C_ + c] = f2bf(xv * sr[c]);
        xbd[h * C_ + c] = f2bf(dv * sd[c]);
        xbf[h * C_ + c] = f2bf(xv * dv);
    }
    __syncthreads();

    const unsigned short* wb = (const unsigned short*)(ws + OFF_WB);
    float* qkv = ws + OFF_QKV + (size_t)b * QKV_PITCH;

    int wave = t >> 5, lane = t & 31;
    int m = lane & 15, hi = lane >> 4, hi8 = hi * 8;

    for (int tt = 0; tt < 16; ++tt) {                       // 8 waves * 16 = 128 tiles
        int tile = wave * 16 + tt;
        int mt = tile >> 2, nt = tile & 3;                  // 32 M-tiles x 4 N-tiles
        int obase = mt * 16;
        const unsigned short* act  = (obase < 128) ? xbr : (obase < 256) ? xbd : xbf;
        const unsigned short* arow = wb + (size_t)(obase + m) * C_;
        const unsigned short* brow = act + (size_t)(nt * 16 + m) * C_;
        floatx8 acc = {};
        for (int k0 = 0; k0 < C_; k0 += 32) {
            ushort16 au, bu;
#pragma unroll
            for (int e = 0; e < 16; ++e)
                au[e] = arow[k0 + hi * 16 + e];
            bu = *(const ushort16*)(brow + k0 + hi * 16);
            acc = __builtin_amdgcn_wmma_f32_16x16x32_bf16(
                false, __builtin_bit_cast(bf16x16, au),
                false, __builtin_bit_cast(bf16x16, bu),
                (short)0, acc, false, false);
        }
#pragma unroll
        for (int r = 0; r < 8; ++r) {
            int o = obase + r + hi8;
            int h = nt * 16 + m;
            if (h < H_) qkv[o * H_ + h] = acc[r];
        }
    }
}

// ---------------------------------------------------- per-channel BN stats q/k/v
__global__ void k_qkvstat(float* ws) {
    __shared__ float red[256];
    int ch = blockIdx.x;                       // 512 channels
    float s = 0.f, q = 0.f;
    for (int idx = threadIdx.x; idx < B_ * H_; idx += 256) {
        int b = idx / H_, h = idx % H_;
        float v = ws[OFF_QKV + (size_t)b * QKV_PITCH + ch * H_ + h];
        s += v; q += v * v;
    }
    s = blk_sum(s, red);
    q = blk_sum(q, red);
    if (threadIdx.x == 0) {
        constexpr float inv = 1.0f / ((float)B_ * H_);
        float mn = s * inv, var = q * inv - mn * mn;
        ws[OFF_QSTAT + ch * 2]     = mn;
        ws[OFF_QSTAT + ch * 2 + 1] = rsqrtf(var + EPSF);
    }
}

// ------------------------------------------------------------------ attention
// PASS 0: qk (WMMA) + qr/kr, accumulate bn_sim stats.
// PASS 1: recompute, BN, softmax, sve (VALU) + sv (WMMA), write so + bn_out stats.
template <int PASS>
__global__ void k_attn(const float* __restrict__ bqg, const float* __restrict__ bqb,
                       const float* __restrict__ bkg, const float* __restrict__ bkb,
                       const float* __restrict__ bvg, const float* __restrict__ bvb,
                       const float* __restrict__ bsg, const float* __restrict__ bsb,
                       const float* __restrict__ rel, float* ws) {
    __shared__ float          qf[16 * H_], kf[16 * H_];
    __shared__ unsigned short qbT[64 * 32], kbT[64 * 32];   // A/B frags, K padded to 32
    __shared__ unsigned short vb[32 * 64];                  // v bf16 [c][j64]
    __shared__ float          s[64 * 64];                   // sim / scratch
    __shared__ unsigned short sb[64 * 64];                  // sim bf16
    __shared__ float          red[128], osum[128];

    int bid = blockIdx.x, b = bid >> 3, g = bid & 7;
    int t = threadIdx.x;
    int wave = t >> 5, lane = t & 31;
    int m = lane & 15, hi = lane >> 4, hi8 = hi * 8;

    const float* qstat = ws + OFF_QSTAT;
    float*       qkv   = ws + OFF_QKV + (size_t)b * QKV_PITCH;

    for (int i = t; i < 64 * 32; i += 128) { qbT[i] = 0; kbT[i] = 0; }
    for (int i = t; i < 32 * 64; i += 128) vb[i] = 0;
    if (PASS == 1) {
        for (int i = t; i < 64 * 64; i += 128) sb[i] = 0;
        if (t < 128) osum[t] = 0.f;
    }
    __syncthreads();

    for (int idx = t; idx < 16 * H_; idx += 128) {          // load + BN q,k
        int c = idx / H_, i = idx % H_;
        {
            int o = g * 16 + c;
            float mn = qstat[o * 2], rs = qstat[o * 2 + 1];
            float v = bqg[o] * rs * (qkv[o * H_ + i] - mn) + bqb[o];
            qf[c * H_ + i] = v; qbT[i * 32 + c] = f2bf(v);
        }
        {
            int o = 128 + g * 16 + c;
            float mn = qstat[o * 2], rs = qstat[o * 2 + 1];
            float v = bkg[o - 128] * rs * (qkv[o * H_ + i] - mn) + bkb[o - 128];
            kf[c * H_ + i] = v; kbT[i * 32 + c] = f2bf(v);
        }
    }
    if (PASS == 1) {
        for (int idx = t; idx < GP_ * H_; idx += 128) {     // load + BN v
            int c = idx / H_, j = idx % H_;
            int o = 256 + g * GP_ + c;
            float mn = qstat[o * 2], rs = qstat[o * 2 + 1];
            float v = bvg[o - 256] * rs * (qkv[o * H_ + j] - mn) + bvb[o - 256];
            vb[c * 64 + j] = f2bf(v);
        }
    }
    __syncthreads();

    // qk = q^T k : 4 waves x 4 N-tiles, K = 16 (zero-padded to 32)
    for (int nt = 0; nt < 4; ++nt) {
        floatx8 acc = {};
        ushort16 au, bu;
#pragma unroll
        for (int e = 0; e < 16; ++e)
            au[e] = qbT[(wave * 16 + m) * 32 + hi * 16 + e];
        bu = *(const ushort16*)(&kbT[(nt * 16 + m) * 32 + hi * 16]);
        acc = __builtin_amdgcn_wmma_f32_16x16x32_bf16(
            false, __builtin_bit_cast(bf16x16, au),
            false, __builtin_bit_cast(bf16x16, bu),
            (short)0, acc, false, false);
#pragma unroll
        for (int r = 0; r < 8; ++r)
            s[(wave * 16 + r + hi8) * 64 + nt * 16 + m] = acc[r];
    }
    __syncthreads();

    if (PASS == 0) {
        float a0 = 0, a1 = 0, a2 = 0, a3 = 0, a4 = 0, a5 = 0;
        for (int idx = t; idx < H_ * H_; idx += 128) {
            int i = idx / H_, j = idx % H_;
            float qk = s[i * 64 + j];
            float qr = 0.f, kr = 0.f;
#pragma unroll
            for (int c = 0; c < 16; ++c) {
                qr += qf[c * H_ + i] * rel[c * 111 + (i - j + 55)];
                kr += kf[c * H_ + j] * rel[(16 + c) * 111 + (j - i + 55)];
            }
            a0 += qk; a1 += qk * qk;
            a2 += qr; a3 += qr * qr;
            a4 += kr; a5 += kr * kr;
        }
        float* ss = ws + OFF_SSTAT;
        a0 = blk_sum(a0, red); a1 = blk_sum(a1, red);
        a2 = blk_sum(a2, red); a3 = blk_sum(a3, red);
        a4 = blk_sum(a4, red); a5 = blk_sum(a5, red);
        if (t == 0) {
            atomicAdd(&ss[g * 2],            a0); atomicAdd(&ss[g * 2 + 1],        a1);
            atomicAdd(&ss[(8 + g) * 2],      a2); atomicAdd(&ss[(8 + g) * 2 + 1],  a3);
            atomicAdd(&ss[(16 + g) * 2],     a4); atomicAdd(&ss[(16 + g) * 2 + 1], a5);
        }
        return;
    }

    // ---------------- PASS 1 ----------------
    const float* ss = ws + OFF_SSTAT;
    constexpr float invS = 1.0f / ((float)B_ * H_ * H_);
    float aK, bK, aQ, bQ, aR, bR;
    {
        float su = ss[g * 2], sq = ss[g * 2 + 1];
        float mn = su * invS, var = sq * invS - mn * mn, rs = rsqrtf(var + EPSF);
        aK = bsg[g] * rs; bK = bsb[g] - mn * aK;
    }
    {
        int ch = 8 + g;
        float su = ss[ch * 2], sq = ss[ch * 2 + 1];
        float mn = su * invS, var = sq * invS - mn * mn, rs = rsqrtf(var + EPSF);
        aQ = bsg[ch] * rs; bQ = bsb[ch] - mn * aQ;
    }
    {
        int ch = 16 + g;
        float su = ss[ch * 2], sq = ss[ch * 2 + 1];
        float mn = su * invS, var = sq * invS - mn * mn, rs = rsqrtf(var + EPSF);
        aR = bsg[ch] * rs; bR = bsb[ch] - mn * aR;
    }
    for (int idx = t; idx < H_ * H_; idx += 128) {
        int i = idx / H_, j = idx % H_;
        float qk = s[i * 64 + j];
        float qr = 0.f, kr = 0.f;
#pragma unroll
        for (int c = 0; c < 16; ++c) {
            qr += qf[c * H_ + i] * rel[c * 111 + (i - j + 55)];
            kr += kf[c * H_ + j] * rel[(16 + c) * 111 + (j - i + 55)];
        }
        s[i * 64 + j] = aK * qk + bK + aQ * qr + bQ + aR * kr + bR;
    }
    __syncthreads();

    if (t < H_) {                                           // row softmax over j
        int i = t;
        float mx = -1e30f;
        for (int j = 0; j < H_; ++j) mx = fmaxf(mx, s[i * 64 + j]);
        float sum = 0.f;
        for (int j = 0; j < H_; ++j) { float e = __expf(s[i * 64 + j] - mx); s[i * 64 + j] = e; sum += e; }
        float inv = 1.0f / sum;
        for (int j = 0; j < H_; ++j) { float p = s[i * 64 + j] * inv; s[i * 64 + j] = p; sb[i * 64 + j] = f2bf(p); }
    }
    __syncthreads();

    float* so    = ws + OFF_SO + (size_t)b * QKV_PITCH;
    float* ostat = ws + OFF_OSTAT;

    for (int idx = t; idx < GP_ * H_; idx += 128) {         // sve (VALU, per-i dots)
        int c = idx / H_, i = idx % H_;
        float v = 0.f;
        for (int j = 0; j < H_; ++j)
            v += s[i * 64 + j] * rel[(32 + c) * 111 + (i - j + 55)];
        so[(2 * (GP_ * g + c) + 1) * H_ + i] = v;
        atomicAdd(&osum[(2 * c + 1) * 2],     v);
        atomicAdd(&osum[(2 * c + 1) * 2 + 1], v * v);
    }
    __syncthreads();

    // sv = sim @ v^T via WMMA: 4 waves (M-tiles) x 2 N-tiles, K = 64 (padded)
    for (int nt = 0; nt < 2; ++nt) {
        floatx8 acc = {};
        for (int k0 = 0; k0 < 64; k0 += 32) {
            ushort16 au, bu;
#pragma unroll
            for (int e = 0; e < 16; ++e)
                au[e] = sb[(wave * 16 + m) * 64 + k0 + hi * 16 + e];
            bu = *(const ushort16*)(&vb[(nt * 16 + m) * 64 + k0 + hi * 16]);
            acc = __builtin_amdgcn_wmma_f32_16x16x32_bf16(
                false, __builtin_bit_cast(bf16x16, au),
                false, __builtin_bit_cast(bf16x16, bu),
                (short)0, acc, false, false);
        }
#pragma unroll
        for (int r = 0; r < 8; ++r)
            s[(wave * 16 + r + hi8) * 64 + nt * 16 + m] = acc[r];   // s <- sv [i][c]
    }
    __syncthreads();

    for (int idx = t; idx < GP_ * H_; idx += 128) {         // write sv channels
        int c = idx / H_, i = idx % H_;
        float v = s[i * 64 + c];
        so[(2 * (GP_ * g + c)) * H_ + i] = v;
        atomicAdd(&osum[(2 * c) * 2],     v);
        atomicAdd(&osum[(2 * c) * 2 + 1], v * v);
    }
    __syncthreads();
    if (t < 128) atomicAdd(&ostat[128 * g + t], osum[t]);   // flush block stats
}

// ---------------------------------------------------------------- finalize
__global__ void k_final(const float* __restrict__ bog, const float* __restrict__ bob,
                        const float* __restrict__ ws, float* __restrict__ out) {
    int n = blockIdx.x >> 8, o = blockIdx.x & 255;
    const float* ostat = ws + OFF_OSTAT;
    constexpr float inv = 1.0f / ((float)B_ * H_);
    int c0 = 2 * o, c1 = 2 * o + 1;
    float m0 = ostat[c0 * 2] * inv, v0 = ostat[c0 * 2 + 1] * inv - m0 * m0;
    float r0 = rsqrtf(v0 + EPSF);
    float A0 = bog[c0] * r0, B0 = bob[c0] - m0 * A0;
    float m1 = ostat[c1 * 2] * inv, v1 = ostat[c1 * 2 + 1] * inv - m1 * m1;
    float r1 = rsqrtf(v1 + EPSF);
    float A1 = bog[c1] * r1, B1 = bob[c1] - m1 * A1;
    for (int idx = threadIdx.x; idx < HW_; idx += 256) {
        int h = idx / W_, w = idx % W_;
        int b = n * W_ + w;
        const float* so = ws + OFF_SO + (size_t)b * QKV_PITCH;
        out[(((size_t)(n * OUT_ + o)) * H_ + h) * W_ + w] =
            A0 * so[c0 * H_ + h] + B0 + A1 * so[c1 * H_ + h] + B1;
    }
}

extern "C" void kernel_launch(void* const* d_in, const int* in_sizes, int n_in,
                              void* d_out, int out_size, void* d_ws, size_t ws_size,
                              hipStream_t stream) {
    const float* x    = (const float*)d_in[0];
    const float* xd   = (const float*)d_in[1];
    const float* cawr = (const float*)d_in[2];
    const float* cabr = (const float*)d_in[3];
    const float* cawd = (const float*)d_in[4];
    const float* cabd = (const float*)d_in[5];
    const float* qw   = (const float*)d_in[6];
    const float* kw   = (const float*)d_in[7];
    const float* vw   = (const float*)d_in[8];
    const float* bqg  = (const float*)d_in[9];
    const float* bqb  = (const float*)d_in[10];
    const float* bkg  = (const float*)d_in[11];
    const float* bkb  = (const float*)d_in[12];
    const float* bvg  = (const float*)d_in[13];
    const float* bvb  = (const float*)d_in[14];
    const float* bsg  = (const float*)d_in[15];
    const float* bsb  = (const float*)d_in[16];
    const float* bog  = (const float*)d_in[17];
    const float* bob  = (const float*)d_in[18];
    const float* rel  = (const float*)d_in[19];
    float* ws  = (float*)d_ws;
    float* out = (float*)d_out;

    k_zero <<<1,    256, 0, stream>>>(ws);
    k_pool <<<N_ * C_, 256, 0, stream>>>(x, xd, ws);
    k_chatt<<<N_,   256, 0, stream>>>(cawr, cabr, cawd, cabd, ws);
    k_wcvt <<<512,  256, 0, stream>>>(qw, kw, vw, ws);
    k_proj <<<B_,   256, 3 * 64 * C_ * sizeof(unsigned short), stream>>>(x, xd, ws);
    k_qkvstat<<<512, 256, 0, stream>>>(ws);
    k_attn<0><<<B_ * G_, 128, 0, stream>>>(bqg, bqb, bkg, bkb, bvg, bvb, bsg, bsb, rel, ws);
    k_attn<1><<<B_ * G_, 128, 0, stream>>>(bqg, bqb, bkg, bkb, bvg, bvb, bsg, bsb, rel, ws);
    k_final<<<N_ * OUT_, 256, 0, stream>>>(bog, bob, ws, out);
}
